// ChannelAttentionModule_40286793237105
// MI455X (gfx1250) — matrix operand order repaired and checked by reading, not compiled, not run on hardware
//
#include <hip/hip_runtime.h>
#include <hip/hip_bf16.h>

// ---------------------------------------------------------------------------
// Channel attention, CDNA5 (gfx1250) WMMA bf16 pipeline.
//   k1: q,k = W*x+b (row-major bf16), v stored transposed vT[n][d] (bf16)
//   k2: attn = q @ k^T  (f32 acc, v_wmma_f32_16x16x32_bf16, double-buffered LDS)
//   k3: row softmax (f32 in, bf16 out)
//   k4: out = attn @ v + x (f32 out, double-buffered LDS, staged epilogue)
// ---------------------------------------------------------------------------

#define C_DIM 512
#define N_DIM 4096
#define B_DIM 8
#define BT 64     // k1 tile edge
#define BK 32     // K step (WMMA bf16 K)
#define MT 128    // k2/k4 tile edge

typedef __attribute__((ext_vector_type(16))) __bf16 bf16x16;
typedef __attribute__((ext_vector_type(8)))  float  f32x8;

struct alignas(16) U128 { unsigned int a, b, c, d; };
struct alignas(8)  U64  { unsigned int a, b; };

union FragU {
    bf16x16        v;
    U128           q[2];
    unsigned short u[16];
};

__device__ __forceinline__ unsigned short f2bf(float f) {
    unsigned int u = __float_as_uint(f);
    unsigned int r = u + 0x7FFFu + ((u >> 16) & 1u);   // round-to-nearest-even
    return (unsigned short)(r >> 16);
}

// ---------------------------------------------------------------------------
// Kernel 1: fused QKV projection.  One workgroup = 64(o) x 64(n) tile,
// X-tile loaded once and reused by wq/wk/wv A-tiles (12 WMMA per K-step/wave).
// ---------------------------------------------------------------------------
__global__ __launch_bounds__(128) void qkv_proj_kernel(
    const float* __restrict__ x,
    const float* __restrict__ wq, const float* __restrict__ bq,
    const float* __restrict__ wk, const float* __restrict__ bk,
    const float* __restrict__ wv, const float* __restrict__ bv,
    unsigned short* __restrict__ qb,    // [B][C][N] bf16
    unsigned short* __restrict__ kb,    // [B][C][N] bf16
    unsigned short* __restrict__ vtb)   // [B][N][C] bf16 (transposed V)
{
    __shared__ union {
        struct {
            unsigned short xt[BT][BK];     // B operand: [n][k] (k contiguous)
            unsigned short wt[3][BT][BK];  // A operand: [o][k] row-major
        } tiles;                           // 16 KB
        unsigned short stage[BT][BT];      // 8 KB epilogue staging
    } sm;

    const int b    = blockIdx.z;
    const int ob   = blockIdx.y * BT;   // output-channel base
    const int nb   = blockIdx.x * BT;   // spatial base
    const int tid  = threadIdx.x;
    const int wave = tid >> 5;          // 0..3
    const int lane = tid & 31;
    const int lh   = lane >> 4;         // lane half (0/1)
    const int ll   = lane & 15;

    const float* wptr[3] = { wq, wk, wv };
    const float* bptr[3] = { bq, bk, bv };

    f32x8 acc[3][4] = {};               // 12 accumulator tiles (q,k,v x 4 n-subtiles)

    const float* xbase = x + (size_t)b * C_DIM * N_DIM + nb;

    for (int ks = 0; ks < C_DIM; ks += BK) {
        __syncthreads();
        // --- stage X tile: 32(k) x 64(n) fp32 -> bf16, transposed to [n][k]
        #pragma unroll
        for (int i = 0; i < 4; ++i) {
            int idx = tid + i * 128;            // 0..511
            int kk  = idx >> 4;                 // 0..31
            int ng  = (idx & 15) << 2;          // 0..60
            const float4 f = *(const float4*)(xbase + (size_t)(ks + kk) * N_DIM + ng);
            if (ks + BK < C_DIM)
                __builtin_prefetch(xbase + (size_t)(ks + BK + kk) * N_DIM + ng, 0, 1);
            sm.tiles.xt[ng + 0][kk] = f2bf(f.x);
            sm.tiles.xt[ng + 1][kk] = f2bf(f.y);
            sm.tiles.xt[ng + 2][kk] = f2bf(f.z);
            sm.tiles.xt[ng + 3][kk] = f2bf(f.w);
        }
        // --- stage W tiles: 64(o) x 32(k) fp32 -> bf16 row-major, x3 weights
        #pragma unroll
        for (int t = 0; t < 3; ++t) {
            const float* wb = wptr[t] + (size_t)ob * C_DIM + ks;
            #pragma unroll
            for (int i = 0; i < 4; ++i) {
                int idx = tid + i * 128;        // 0..511
                int o   = idx >> 3;             // 0..63
                int kg  = (idx & 7) << 2;       // 0..28
                const float4 f = *(const float4*)(wb + (size_t)o * C_DIM + kg);
                U64 p;
                p.a = ((unsigned)f2bf(f.y) << 16) | f2bf(f.x);
                p.b = ((unsigned)f2bf(f.w) << 16) | f2bf(f.z);
                *(U64*)&sm.tiles.wt[t][o][kg] = p;
            }
        }
        __syncthreads();

        // --- B fragments: lane = column n, K 8h..8h+7 and 16+8h..16+8h+7
        FragU bfr[4];
        #pragma unroll
        for (int j = 0; j < 4; ++j) {
            const unsigned short* p = &sm.tiles.xt[j * 16 + ll][lh * 8];
            bfr[j].q[0] = *(const U128*)p;
            bfr[j].q[1] = *(const U128*)(p + 16);
        }
        // --- A fragments per weight + 12 WMMAs
        #pragma unroll
        for (int t = 0; t < 3; ++t) {
            FragU af;
            const unsigned short* p = &sm.tiles.wt[t][wave * 16 + ll][lh * 8];
            af.q[0] = *(const U128*)p;          // K 0..7  (or 8..15)
            af.q[1] = *(const U128*)(p + 16);   // K 16..23 (or 24..31)
            #pragma unroll
            for (int j = 0; j < 4; ++j)
                acc[t][j] = __builtin_amdgcn_wmma_f32_16x16x32_bf16(
                    false, af.v, false, bfr[j].v, (short)0, acc[t][j], false, false);
        }
    }

    // ------------------------ epilogue: bias + stores ------------------------
    #pragma unroll
    for (int t = 0; t < 3; ++t) {
        float bias[8];
        #pragma unroll
        for (int r = 0; r < 8; ++r)
            bias[r] = bptr[t][ob + wave * 16 + lh * 8 + r];

        if (t == 2) {
            // V: accumulator lane holds a column fragment -> contiguous 16B
            // store directly into vT[b][n][o..o+7].
            #pragma unroll
            for (int j = 0; j < 4; ++j) {
                union { U128 q; unsigned short s[8]; } pk;
                #pragma unroll
                for (int r = 0; r < 8; ++r) pk.s[r] = f2bf(acc[2][j][r] + bias[r]);
                int n = nb + j * 16 + ll;
                int o = ob + wave * 16 + lh * 8;
                *(U128*)(vtb + (size_t)b * N_DIM * C_DIM + (size_t)n * C_DIM + o) = pk.q;
            }
        } else {
            // Q/K: stage through LDS so global bf16 stores are coalesced rows.
            __syncthreads();
            #pragma unroll
            for (int j = 0; j < 4; ++j)
                #pragma unroll
                for (int r = 0; r < 8; ++r)
                    sm.stage[wave * 16 + lh * 8 + r][j * 16 + ll] =
                        f2bf(acc[t][j][r] + bias[r]);
            __syncthreads();
            unsigned short* dst = (t == 0 ? qb : kb) + (size_t)b * C_DIM * N_DIM;
            #pragma unroll
            for (int i = 0; i < 4; ++i) {
                int idx = tid + i * 128;        // 0..511
                int row = idx >> 3;             // 0..63
                int cg  = (idx & 7) << 3;       // 0..56
                *(U128*)(dst + (size_t)(ob + row) * N_DIM + nb + cg) =
                    *(U128*)&sm.stage[row][cg];
            }
        }
    }
}

// ---------------------------------------------------------------------------
// Kernel 2: attn[b] = Q (C x N) @ K^T  -> f32 [B][C][C]
// 256 threads, 128x128 tile, double-buffered LDS, 8 WMMA per wave per K-step.
// ---------------------------------------------------------------------------
__global__ __launch_bounds__(256) void attn_gemm_kernel(
    const unsigned short* __restrict__ qb,
    const unsigned short* __restrict__ kb,
    float* __restrict__ attn)
{
    __shared__ unsigned short at[2][MT][BK];   // A tiles (rows of Q), 16 KB
    __shared__ unsigned short bt[2][MT][BK];   // B tiles (rows of K), 16 KB

    const int b    = blockIdx.z;
    const int cb   = blockIdx.y * MT;
    const int db   = blockIdx.x * MT;
    const int tid  = threadIdx.x;
    const int wave = tid >> 5;                 // 0..7
    const int lane = tid & 31;
    const int lh   = lane >> 4;
    const int ll   = lane & 15;
    const int mrow = (wave & 3) * 32;          // wave row base in tile
    const int ncol = (wave >> 2) * 64;         // wave col base in tile

    const int crow  = tid >> 1;                // copy row 0..127
    const int chalf = (tid & 1) * 16;

    const unsigned short* qrow =
        qb + (size_t)b * C_DIM * N_DIM + (size_t)(cb + crow) * N_DIM + chalf;
    const unsigned short* krow =
        kb + (size_t)b * C_DIM * N_DIM + (size_t)(db + crow) * N_DIM + chalf;

    f32x8 acc[2][4] = {};

    const int STEPS = N_DIM / BK;              // 128
    U128 pa0, pa1, pb0, pb1;
    pa0 = *(const U128*)qrow;       pa1 = *(const U128*)(qrow + 8);
    pb0 = *(const U128*)krow;       pb1 = *(const U128*)(krow + 8);
    *(U128*)&at[0][crow][chalf]     = pa0;
    *(U128*)&at[0][crow][chalf + 8] = pa1;
    *(U128*)&bt[0][crow][chalf]     = pb0;
    *(U128*)&bt[0][crow][chalf + 8] = pb1;

    for (int s = 0; s < STEPS; ++s) {
        __syncthreads();
        const int cur = s & 1, nxt = cur ^ 1;
        if (s + 1 < STEPS) {                   // global prefetch into registers
            const unsigned short* qn = qrow + (size_t)(s + 1) * BK;
            const unsigned short* kn = krow + (size_t)(s + 1) * BK;
            pa0 = *(const U128*)qn; pa1 = *(const U128*)(qn + 8);
            pb0 = *(const U128*)kn; pb1 = *(const U128*)(kn + 8);
        }
        if (s + 2 < STEPS) {                   // L2 prefetch two steps ahead
            __builtin_prefetch(qrow + (size_t)(s + 2) * BK, 0, 1);
            __builtin_prefetch(krow + (size_t)(s + 2) * BK, 0, 1);
        }

        FragU af[2];
        #pragma unroll
        for (int ms = 0; ms < 2; ++ms) {
            const unsigned short* p = &at[cur][mrow + ms * 16 + ll][lh * 8];
            af[ms].q[0] = *(const U128*)p;
            af[ms].q[1] = *(const U128*)(p + 16);
        }
        #pragma unroll
        for (int j = 0; j < 4; ++j) {
            FragU bfr;
            const unsigned short* bp = &bt[cur][ncol + j * 16 + ll][lh * 8];
            bfr.q[0] = *(const U128*)bp;
            bfr.q[1] = *(const U128*)(bp + 16);
            acc[0][j] = __builtin_amdgcn_wmma_f32_16x16x32_bf16(
                false, af[0].v, false, bfr.v, (short)0, acc[0][j], false, false);
            acc[1][j] = __builtin_amdgcn_wmma_f32_16x16x32_bf16(
                false, af[1].v, false, bfr.v, (short)0, acc[1][j], false, false);
        }

        if (s + 1 < STEPS) {                   // fill next buffer (race-free)
            *(U128*)&at[nxt][crow][chalf]     = pa0;
            *(U128*)&at[nxt][crow][chalf + 8] = pa1;
            *(U128*)&bt[nxt][crow][chalf]     = pb0;
            *(U128*)&bt[nxt][crow][chalf + 8] = pb1;
        }
    }

    float* arow = attn + (size_t)b * C_DIM * C_DIM;
    #pragma unroll
    for (int ms = 0; ms < 2; ++ms)
        #pragma unroll
        for (int j = 0; j < 4; ++j)
            #pragma unroll
            for (int r = 0; r < 8; ++r)
                arow[(size_t)(cb + mrow + ms * 16 + lh * 8 + r) * C_DIM
                     + db + ncol + j * 16 + ll] = acc[ms][j][r];
}

// ---------------------------------------------------------------------------
// Kernel 3: row softmax over last dim (512), f32 -> bf16
// ---------------------------------------------------------------------------
__global__ __launch_bounds__(128) void softmax_kernel(
    const float* __restrict__ attn, unsigned short* __restrict__ attnb)
{
    __shared__ float red[128];
    const int row = blockIdx.x;                    // 0 .. B*C-1
    const int t   = threadIdx.x;
    const float* a = attn + (size_t)row * C_DIM;

    float v[4];
    float m = -3.4e38f;
    #pragma unroll
    for (int i = 0; i < 4; ++i) { v[i] = a[t + i * 128]; m = fmaxf(m, v[i]); }
    red[t] = m; __syncthreads();
    for (int s = 64; s > 0; s >>= 1) {
        if (t < s) red[t] = fmaxf(red[t], red[t + s]);
        __syncthreads();
    }
    m = red[0]; __syncthreads();

    float sum = 0.f;
    #pragma unroll
    for (int i = 0; i < 4; ++i) { v[i] = __expf(v[i] - m); sum += v[i]; }
    red[t] = sum; __syncthreads();
    for (int s = 64; s > 0; s >>= 1) {
        if (t < s) red[t] += red[t + s];
        __syncthreads();
    }
    const float inv = 1.f / red[0];

    unsigned short* o = attnb + (size_t)row * C_DIM;
    #pragma unroll
    for (int i = 0; i < 4; ++i) o[t + i * 128] = f2bf(v[i] * inv);
}

// ---------------------------------------------------------------------------
// Kernel 4: out = attn (C x C) @ V (C x N) + x  -> f32 [B][C][H][W]
// 256 threads, 128x128 tile, double-buffered LDS; staged residual epilogue.
// ---------------------------------------------------------------------------
__global__ __launch_bounds__(256) void out_gemm_kernel(
    const unsigned short* __restrict__ attnb,   // [B][C][C] bf16
    const unsigned short* __restrict__ vtb,     // [B][N][C] bf16
    const float* __restrict__ x,
    float* __restrict__ out)
{
    __shared__ union {
        struct {
            unsigned short at[2][MT][BK];    // 16 KB
            unsigned short bt[2][MT][BK];    // 16 KB
        } tiles;
        float stage[MT / 2][MT];             // 32 KB epilogue staging
    } sm;

    const int b    = blockIdx.z;
    const int cb   = blockIdx.y * MT;
    const int nb   = blockIdx.x * MT;
    const int tid  = threadIdx.x;
    const int wave = tid >> 5;
    const int lane = tid & 31;
    const int lh   = lane >> 4;
    const int ll   = lane & 15;
    const int mrow = (wave & 3) * 32;
    const int ncol = (wave >> 2) * 64;

    const int crow  = tid >> 1;
    const int chalf = (tid & 1) * 16;

    const unsigned short* arow =
        attnb + (size_t)b * C_DIM * C_DIM + (size_t)(cb + crow) * C_DIM + chalf;
    const unsigned short* vrow =
        vtb + (size_t)b * N_DIM * C_DIM + (size_t)(nb + crow) * C_DIM + chalf;

    f32x8 acc[2][4] = {};

    const int STEPS = C_DIM / BK;              // 16
    U128 pa0, pa1, pb0, pb1;
    pa0 = *(const U128*)arow;       pa1 = *(const U128*)(arow + 8);
    pb0 = *(const U128*)vrow;       pb1 = *(const U128*)(vrow + 8);
    *(U128*)&sm.tiles.at[0][crow][chalf]     = pa0;
    *(U128*)&sm.tiles.at[0][crow][chalf + 8] = pa1;
    *(U128*)&sm.tiles.bt[0][crow][chalf]     = pb0;
    *(U128*)&sm.tiles.bt[0][crow][chalf + 8] = pb1;

    for (int s = 0; s < STEPS; ++s) {
        __syncthreads();
        const int cur = s & 1, nxt = cur ^ 1;
        if (s + 1 < STEPS) {
            const unsigned short* an = arow + (size_t)(s + 1) * BK;
            const unsigned short* vn = vrow + (size_t)(s + 1) * BK;
            pa0 = *(const U128*)an; pa1 = *(const U128*)(an + 8);
            pb0 = *(const U128*)vn; pb1 = *(const U128*)(vn + 8);
        }
        if (s + 2 < STEPS) {
            __builtin_prefetch(arow + (size_t)(s + 2) * BK, 0, 1);
            __builtin_prefetch(vrow + (size_t)(s + 2) * BK, 0, 1);
        }

        FragU af[2];
        #pragma unroll
        for (int ms = 0; ms < 2; ++ms) {
            const unsigned short* p = &sm.tiles.at[cur][mrow + ms * 16 + ll][lh * 8];
            af[ms].q[0] = *(const U128*)p;
            af[ms].q[1] = *(const U128*)(p + 16);
        }
        #pragma unroll
        for (int j = 0; j < 4; ++j) {
            FragU bfr;
            const unsigned short* bp = &sm.tiles.bt[cur][ncol + j * 16 + ll][lh * 8];
            bfr.q[0] = *(const U128*)bp;
            bfr.q[1] = *(const U128*)(bp + 16);
            acc[0][j] = __builtin_amdgcn_wmma_f32_16x16x32_bf16(
                false, af[0].v, false, bfr.v, (short)0, acc[0][j], false, false);
            acc[1][j] = __builtin_amdgcn_wmma_f32_16x16x32_bf16(
                false, af[1].v, false, bfr.v, (short)0, acc[1][j], false, false);
        }

        if (s + 1 < STEPS) {
            *(U128*)&sm.tiles.at[nxt][crow][chalf]     = pa0;
            *(U128*)&sm.tiles.at[nxt][crow][chalf + 8] = pa1;
            *(U128*)&sm.tiles.bt[nxt][crow][chalf]     = pb0;
            *(U128*)&sm.tiles.bt[nxt][crow][chalf + 8] = pb1;
        }
    }

    // Epilogue: stage 64 rows at a time (f32), fuse +x residual, float4 stores.
    const float* xr   = x   + (size_t)b * C_DIM * N_DIM;
    float*       orow = out + (size_t)b * C_DIM * N_DIM;
    #pragma unroll
    for (int half = 0; half < 2; ++half) {
        __syncthreads();
        if (((wave & 3) >> 1) == half) {
            const int srow = mrow - half * 64;     // 0 or 32
            #pragma unroll
            for (int ms = 0; ms < 2; ++ms)
                #pragma unroll
                for (int j = 0; j < 4; ++j)
                    #pragma unroll
                    for (int r = 0; r < 8; ++r)
                        sm.stage[srow + ms * 16 + lh * 8 + r][ncol + j * 16 + ll] =
                            acc[ms][j][r];
        }
        __syncthreads();
        #pragma unroll
        for (int i = 0; i < 8; ++i) {
            int idx = tid + i * 256;       // 0..2047
            int row = idx >> 5;            // 0..63
            int cg  = (idx & 31) << 2;     // 0..124
            float4 sv = *(float4*)&sm.stage[row][cg];
            size_t g  = (size_t)(cb + half * 64 + row) * N_DIM + nb + cg;
            float4 xv = *(const float4*)(xr + g);
            sv.x += xv.x; sv.y += xv.y; sv.z += xv.z; sv.w += xv.w;
            *(float4*)(orow + g) = sv;
        }
    }
}

// ---------------------------------------------------------------------------
extern "C" void kernel_launch(void* const* d_in, const int* in_sizes, int n_in,
                              void* d_out, int out_size, void* d_ws, size_t ws_size,
                              hipStream_t stream) {
    (void)in_sizes; (void)n_in; (void)out_size; (void)ws_size;

    const float* x  = (const float*)d_in[0];
    const float* wq = (const float*)d_in[1];
    const float* bq = (const float*)d_in[2];
    const float* wk = (const float*)d_in[3];
    const float* bk = (const float*)d_in[4];
    const float* wv = (const float*)d_in[5];
    const float* bv = (const float*)d_in[6];

    // workspace layout (bf16 unless noted)
    const size_t qkvN = (size_t)B_DIM * C_DIM * N_DIM;   // 16.78M elems
    unsigned short* qb    = (unsigned short*)d_ws;        // [B][C][N]
    unsigned short* kb    = qb  + qkvN;                   // [B][C][N]
    unsigned short* vtb   = kb  + qkvN;                   // [B][N][C]
    float*          attn  = (float*)(vtb + qkvN);         // [B][C][C] f32
    unsigned short* attnb = (unsigned short*)(attn + (size_t)B_DIM * C_DIM * C_DIM);

    qkv_proj_kernel<<<dim3(N_DIM / BT, C_DIM / BT, B_DIM), 128, 0, stream>>>(
        x, wq, bq, wk, bk, wv, bv, qb, kb, vtb);

    attn_gemm_kernel<<<dim3(C_DIM / MT, C_DIM / MT, B_DIM), 256, 0, stream>>>(
        qb, kb, attn);

    softmax_kernel<<<dim3(B_DIM * C_DIM), 128, 0, stream>>>(attn, attnb);

    out_gemm_kernel<<<dim3(N_DIM / MT, C_DIM / MT, B_DIM), 256, 0, stream>>>(
        attnb, vtb, x, (float*)d_out);
}
